// ResidualQuantizer_68332929679829
// MI455X (gfx1250) — matrix operand order repaired and checked by reading, not compiled, not run on hardware
//
#include <hip/hip_runtime.h>
#include <stdint.h>

// Residual VQ (4 stages, 2048 codes, dim 128) for MI455X / gfx1250.
// Scores via v_wmma_f32_16x16x32_bf16; codebook tiles staged into LDS by the
// Tensor Data Mover (tensor_load_to_lds + s_wait_tensorcnt), double-buffered.

#define NPTS   131072
#define DIM    128
#define NCB    4
#define CBS    2048
#define COMMIT 0.25f

#define TILE_CODES 64
#define LDS_ROWW   68                      // 64 kp-words + 4 pad (bank spread)
#define TILEW      (TILE_CODES * LDS_ROWW) // u32 words per buffer

typedef __attribute__((ext_vector_type(16))) __bf16   v16bf;
typedef __attribute__((ext_vector_type(8)))  float    v8f;
typedef __attribute__((ext_vector_type(8)))  uint32_t v8u;
typedef __attribute__((ext_vector_type(4)))  uint32_t v4u;
typedef __attribute__((ext_vector_type(8)))  int      v8i;
typedef __attribute__((ext_vector_type(4)))  int      v4i;

__device__ __forceinline__ uint32_t f2bf(float f) {
  uint32_t u = __builtin_bit_cast(uint32_t, f);
  return (u + 0x7FFFu + ((u >> 16) & 1u)) >> 16;   // round-to-nearest-even
}
__device__ __forceinline__ uint32_t pkbf(float lo, float hi) {
  return f2bf(lo) | (f2bf(hi) << 16);
}

// ---------------------------------------------------------------------------
// TDM: load one 64-row x 256B codebook tile into LDS with a 68-word row
// stride via the pad fields (64 DWORDs -> pad 4 DWORDs).  Issued per-wave.
// ---------------------------------------------------------------------------
#if __has_builtin(__builtin_amdgcn_tensor_load_to_lds)
#define HAVE_TDM 1
__device__ __forceinline__ void tdm_load_tile(const uint32_t* gsrc,
                                              uint32_t lds_byte_off) {
  const uint64_t ga = (uint64_t)(uintptr_t)gsrc;
  v4u g0;
  g0[0] = 1u;                                  // count=1, user descriptor
  g0[1] = lds_byte_off;                        // lds_addr (bytes)
  g0[2] = (uint32_t)ga;                        // global_addr[31:0]
  g0[3] = (uint32_t)(ga >> 32) | (2u << 30);   // global_addr[56:32] | type=2
  v8i g1;
  g1[0] = (int)((2u << 16)      // data_size = 4B
              | (1u << 20)      // pad_enable
              | (5u << 22)      // pad_interval: every 64 DWORDs
              | (3u << 25));    // pad_amount: 4 DWORDs
  g1[1] = (int)(64u << 16);     // tensor_dim0 = 64 (low 16 bits)
  g1[2] = (int)(64u << 16);     // tensor_dim0 hi=0 | tensor_dim1 = 64
  g1[3] = (int)(64u << 16);     // tensor_dim1 hi=0 | tile_dim0 = 64
  g1[4] = (int)64u;             // tile_dim1 = 64, tile_dim2 = 0
  g1[5] = (int)64u;             // tensor_dim0_stride = 64 (low 32 bits)
  g1[6] = 0;                    // stride hi | tensor_dim1_stride lo
  g1[7] = 0;
  const v4i gz = {0, 0, 0, 0};
#if __clang_major__ >= 23
  const v8i gz8 = {0, 0, 0, 0, 0, 0, 0, 0};
  __builtin_amdgcn_tensor_load_to_lds(g0, g1, gz, gz, gz8, 0);
#else
  __builtin_amdgcn_tensor_load_to_lds(g0, g1, gz, gz, 0);
#endif
}
#else
#define HAVE_TDM 0
#endif

// ---------------------------------------------------------------------------
// Prep: pack fp32 codebooks into K-pair-packed bf16 rows (WMMA-B friendly)
// and compute bias[c] = -0.5*||c||^2.  One wave per codebook row.
// ---------------------------------------------------------------------------
__global__ __launch_bounds__(256) void rvq_prep_kernel(
    const float* __restrict__ cb, uint32_t* __restrict__ cbP,
    float* __restrict__ bias) {
  const int wid  = blockIdx.x * 8 + (threadIdx.x >> 5);
  const int lane = threadIdx.x & 31;
  const float4 f = *(const float4*)(cb + (size_t)wid * DIM + lane * 4);
  float s = f.x * f.x + f.y * f.y + f.z * f.z + f.w * f.w;
  uint2 pk = make_uint2(pkbf(f.x, f.y), pkbf(f.z, f.w));
  *(uint2*)(cbP + (size_t)wid * 64 + lane * 2) = pk;
#pragma unroll
  for (int off = 16; off; off >>= 1) s += __shfl_xor(s, off);
  if (lane == 0) bias[wid] = -0.5f * s;
}

__global__ void rvq_zero_loss(float* loss) {
  if (threadIdx.x == 0 && blockIdx.x == 0) *loss = 0.0f;
}

// ---------------------------------------------------------------------------
// One RVQ stage. Block = 256 threads = 8 waves; block owns 128 rows, each
// wave 16 rows. 2048 codes in chunks of 64, double-buffered through LDS.
// ---------------------------------------------------------------------------
__global__ __launch_bounds__(256) void rvq_stage_kernel(
    const float* __restrict__ resIn, float* __restrict__ resOut,
    const uint32_t* __restrict__ cbP, const float* __restrict__ bias,
    const float* __restrict__ cbF, float* __restrict__ codesOut,
    const float* __restrict__ zE, float* __restrict__ zQ,
    float* __restrict__ lossAcc, int stage, int isLast) {
  __shared__ uint32_t tile[2 * TILEW];

  const int t    = threadIdx.x;
  const int lane = t & 31;
  const int wave = t >> 5;
  const int n16  = lane & 15;   // A: row M; B/C: column N
  const int kh   = lane >> 4;   // K-half select
  const int rows0 = blockIdx.x * 128 + wave * 16;

#if HAVE_TDM
  const uint32_t ldsBase = (uint32_t)(uintptr_t)(void*)tile;
  // prefetch chunk 0 into buffer 0 (overlaps the A-fragment global loads)
  if (wave == 0) tdm_load_tile(cbP, ldsBase);
#else
#pragma unroll
  for (int q = 0; q < 4; q++) {
    const int idx = q * 256 + t;
    const int crow = idx >> 4, seg = idx & 15;
    *(uint4*)&tile[crow * LDS_ROWW + seg * 4] =
        *(const uint4*)&cbP[(size_t)crow * 64 + seg * 4];
  }
#endif

  // --- A fragments: 16 rows x 128 dims as 4 bf16 16x16x32 fragments -------
  v16bf afr[4];
  {
    const float* rp = resIn + (size_t)(rows0 + n16) * DIM;
#pragma unroll
    for (int kc = 0; kc < 4; kc++) {
      const int base = kc * 32 + kh * 8;
      float4 p0 = *(const float4*)(rp + base);
      float4 p1 = *(const float4*)(rp + base + 4);
      float4 p2 = *(const float4*)(rp + base + 16);
      float4 p3 = *(const float4*)(rp + base + 20);
      v8u w;
      w[0] = pkbf(p0.x, p0.y); w[1] = pkbf(p0.z, p0.w);
      w[2] = pkbf(p1.x, p1.y); w[3] = pkbf(p1.z, p1.w);
      w[4] = pkbf(p2.x, p2.y); w[5] = pkbf(p2.z, p2.w);
      w[6] = pkbf(p3.x, p3.y); w[7] = pkbf(p3.z, p3.w);
      afr[kc] = __builtin_bit_cast(v16bf, w);
    }
  }

#if HAVE_TDM
  if (wave == 0) __builtin_amdgcn_s_wait_tensorcnt(0);
#endif
  __syncthreads();

  float best[8];
  int   bidx[8];
#pragma unroll
  for (int j = 0; j < 8; j++) { best[j] = -3.4e38f; bidx[j] = 0; }

  for (int chunk = 0; chunk < CBS / TILE_CODES; ++chunk) {
    const int c0 = chunk * TILE_CODES;
    const uint32_t* cur = tile + (chunk & 1) * TILEW;

    // prefetch next chunk into the other buffer (read-safe: last read in the
    // previous iteration, end-of-iteration barrier has everyone past it)
    if (chunk + 1 < CBS / TILE_CODES) {
#if HAVE_TDM
      if (wave == 0)
        tdm_load_tile(cbP + (size_t)(c0 + TILE_CODES) * 64,
                      ldsBase + (uint32_t)(((chunk + 1) & 1) * TILEW * 4));
#else
      uint32_t* nxt = tile + ((chunk + 1) & 1) * TILEW;
#pragma unroll
      for (int q = 0; q < 4; q++) {
        const int idx = q * 256 + t;
        const int crow = idx >> 4, seg = idx & 15;
        *(uint4*)&nxt[crow * LDS_ROWW + seg * 4] =
            *(const uint4*)&cbP[(size_t)(c0 + TILE_CODES + crow) * 64 + seg * 4];
      }
#endif
    }

#pragma unroll
    for (int ti = 0; ti < 4; ti++) {
      // hoist all 8 b128 LDS loads for this 16-code tile ahead of the WMMAs
      uint4 bq[8];
#pragma unroll
      for (int kc = 0; kc < 4; kc++) {
        const uint32_t* bp = &cur[(ti * 16 + n16) * LDS_ROWW + kc * 16 + kh * 4];
        bq[2 * kc]     = *(const uint4*)bp;
        bq[2 * kc + 1] = *(const uint4*)(bp + 8);
      }
      v8f acc = {};
#pragma unroll
      for (int kc = 0; kc < 4; kc++) {
        v8u w;
        w[0] = bq[2 * kc].x;     w[1] = bq[2 * kc].y;
        w[2] = bq[2 * kc].z;     w[3] = bq[2 * kc].w;
        w[4] = bq[2 * kc + 1].x; w[5] = bq[2 * kc + 1].y;
        w[6] = bq[2 * kc + 1].z; w[7] = bq[2 * kc + 1].w;
        v16bf bfr = __builtin_bit_cast(v16bf, w);
        acc = __builtin_amdgcn_wmma_f32_16x16x32_bf16(
            false, afr[kc], false, bfr, (short)0, acc, false, false);
      }
      const int   code = c0 + ti * 16 + n16;
      const float bv   = bias[code];
#pragma unroll
      for (int j = 0; j < 8; j++) {
        const float m = acc[j] + bv;
        if (m > best[j]) { best[j] = m; bidx[j] = code; }
      }
    }

#if HAVE_TDM
    if (wave == 0) __builtin_amdgcn_s_wait_tensorcnt(0);
#endif
    __syncthreads();
  }

  // cross-lane argmax over the 16-lane N groups; tie -> lower index
#pragma unroll
  for (int j = 0; j < 8; j++) {
#pragma unroll
    for (int off = 1; off < 16; off <<= 1) {
      const float ov = __shfl_xor(best[j], off);
      const int   oi = __shfl_xor(bidx[j], off);
      if (ov > best[j] || (ov == best[j] && oi < bidx[j])) {
        best[j] = ov; bidx[j] = oi;
      }
    }
  }

  // exact fp32 residual update, codes, (last stage) z_q + loss
  float lsum = 0.0f;
#pragma unroll
  for (int j = 0; j < 8; j++) {
    const int i0 = __shfl(bidx[j], 0);    // row rows0 + j
    const int i1 = __shfl(bidx[j], 16);   // row rows0 + 8 + j
#pragma unroll
    for (int h = 0; h < 2; h++) {
      const int row = rows0 + j + h * 8;
      const int ci  = h ? i1 : i0;
      float4 rv = *(const float4*)(resIn + (size_t)row * DIM + lane * 4);
      float4 cv = *(const float4*)(cbF + (size_t)ci * DIM + lane * 4);
      float4 nr = make_float4(rv.x - cv.x, rv.y - cv.y, rv.z - cv.z, rv.w - cv.w);
      *(float4*)(resOut + (size_t)row * DIM + lane * 4) = nr;
      if (isLast) {
        float4 ze = *(const float4*)(zE + (size_t)row * DIM + lane * 4);
        *(float4*)(zQ + (size_t)row * DIM + lane * 4) =
            make_float4(ze.x - nr.x, ze.y - nr.y, ze.z - nr.z, ze.w - nr.w);
        lsum += nr.x * nr.x + nr.y * nr.y + nr.z * nr.z + nr.w * nr.w;
      }
      if (lane == 0) codesOut[(size_t)row * NCB + stage] = (float)ci;
    }
  }
  if (isLast) {
#pragma unroll
    for (int off = 16; off; off >>= 1) lsum += __shfl_xor(lsum, off);
    if (lane == 0)
      atomicAdd(lossAcc, (1.0f + COMMIT) * lsum / ((float)NPTS * (float)DIM));
  }
}

// ---------------------------------------------------------------------------
extern "C" void kernel_launch(void* const* d_in, const int* in_sizes, int n_in,
                              void* d_out, int out_size, void* d_ws,
                              size_t ws_size, hipStream_t stream) {
  const float* z_e = (const float*)d_in[0];       // [N,128]
  const float* cbs = (const float*)d_in[1];       // [4,2048,128]

  float* zQ    = (float*)d_out;                   // [N,128]
  float* codes = zQ + (size_t)NPTS * DIM;         // [N,4] (as float)
  float* loss  = codes + (size_t)NPTS * NCB;      // [1]

  // workspace: 64MB residual + 2MB packed bf16 codebooks + 32KB bias
  float*    res  = (float*)d_ws;
  uint32_t* cbP  = (uint32_t*)(res + (size_t)NPTS * DIM);
  float*    bias = (float*)(cbP + (size_t)NCB * CBS * 64);

  rvq_prep_kernel<<<NCB * CBS / 8, 256, 0, stream>>>(cbs, cbP, bias);
  rvq_zero_loss<<<1, 32, 0, stream>>>(loss);

  for (int s = 0; s < NCB; ++s) {
    const float* rin = (s == 0) ? z_e : res;
    rvq_stage_kernel<<<NPTS / 128, 256, 0, stream>>>(
        rin, res, cbP + (size_t)s * CBS * 64, bias + (size_t)s * CBS,
        cbs + (size_t)s * CBS * DIM, codes, z_e, zQ, loss, s,
        (s == NCB - 1) ? 1 : 0);
  }
}
